// DCN_1563368096045
// MI455X (gfx1250) — matrix operand order repaired and checked by reading, not compiled, not run on hardware
//
#include <hip/hip_runtime.h>
#include <hip/hip_bf16.h>

// ----------------------------- problem constants -----------------------------
#define LSEQ   200
#define BATCH  2048
#define HID    256
#define HIN    64
#define G4     (4 * HID)
#define NCLUST 64

// ----------------------------- WMMA vector types -----------------------------
typedef __attribute__((ext_vector_type(16))) __bf16 v16bf;
typedef __attribute__((ext_vector_type(8)))  __bf16 v8bf;
typedef __attribute__((ext_vector_type(8)))  float  v8f;

// ----------------------------- helpers ---------------------------------------
__device__ __forceinline__ __bf16 f2bf(float f) {
  unsigned u = __builtin_bit_cast(unsigned, f);
  unsigned r = (u + 0x7FFFu + ((u >> 16) & 1u)) >> 16;   // round-to-nearest-even
  return __builtin_bit_cast(__bf16, (unsigned short)r);
}
__device__ __forceinline__ float sigmoidf_(float x) { return 1.f / (1.f + __expf(-x)); }
__device__ __forceinline__ float tanhf_(float x) {
  x = fminf(fmaxf(x, -15.f), 15.f);
  float e = __expf(-2.f * x);
  return (1.f - e) / (1.f + e);
}
__device__ __forceinline__ float wave_reduce_sum(float v) {
  #pragma unroll
  for (int off = 16; off > 0; off >>= 1) v += __shfl_down(v, off, 32);
  return v;  // valid in lane 0
}

// A-fragment (16x32 bf16, M x K), ISA A-layout (interleaved K):
//   lane<16 : VGPR0-3 = K k0+0..7, VGPR4-7 = K k0+16..23
//   lane>=16: VGPR0-3 = K k0+8..15, VGPR4-7 = K k0+24..31
// Source row-major [row][k], ld multiple of 8, 16B-aligned rows.
__device__ __forceinline__ v16bf load_fragA_bf(const __bf16* __restrict__ base,
                                               int row, int ld, int k0, int half) {
  const __bf16* p = base + (size_t)row * ld + k0 + 8 * half;
  v8bf lo = *(const v8bf*)(p);        // global_load_b128
  v8bf hi = *(const v8bf*)(p + 16);   // global_load_b128
  v16bf r;
  #pragma unroll
  for (int e = 0; e < 8; ++e) { r[e] = lo[e]; r[8 + e] = hi[e]; }
  return r;
}

// Same A-fragment but from fp32 memory with on-the-fly bf16 conversion.
__device__ __forceinline__ v16bf load_fragA_f32(const float* __restrict__ base,
                                                int row, int ld, int k0, int half) {
  const float* p = base + (size_t)row * ld + k0 + 8 * half;
  float4 a0 = *(const float4*)(p);
  float4 a1 = *(const float4*)(p + 4);
  float4 b0 = *(const float4*)(p + 16);
  float4 b1 = *(const float4*)(p + 20);
  v16bf r;
  r[0]=f2bf(a0.x);  r[1]=f2bf(a0.y);  r[2]=f2bf(a0.z);  r[3]=f2bf(a0.w);
  r[4]=f2bf(a1.x);  r[5]=f2bf(a1.y);  r[6]=f2bf(a1.z);  r[7]=f2bf(a1.w);
  r[8]=f2bf(b0.x);  r[9]=f2bf(b0.y);  r[10]=f2bf(b0.z); r[11]=f2bf(b0.w);
  r[12]=f2bf(b1.x); r[13]=f2bf(b1.y); r[14]=f2bf(b1.z); r[15]=f2bf(b1.w);
  return r;
}

// B-fragment (32x16 bf16, K x N), same interleaved K as A-fragment:
//   lane n (0-15): K = k0+0..7, k0+16..23 ; lane n+16: K = k0+8..15, k0+24..31
// B[k][n] = W[n][k]; W row-major [n][k] -> two 16-byte runs per lane.
__device__ __forceinline__ v16bf load_fragB_bf(const __bf16* __restrict__ base,
                                               int row, int ld, int k0, int half) {
  const __bf16* p = base + (size_t)row * ld + k0 + 8 * half;
  v8bf lo = *(const v8bf*)(p);        // K = 8h+0..7
  v8bf hi = *(const v8bf*)(p + 16);   // K = 16+8h+0..7
  v16bf r;
  #pragma unroll
  for (int e = 0; e < 8; ++e) { r[e] = lo[e]; r[8 + e] = hi[e]; }
  return r;
}

// ----------------------------- fused LSTM step -------------------------------
// One kernel = one timestep of one layer.
// grid = (BATCH/64, HID/32), block = 256 (8 waves).
// Each wave owns a 16(batch) x 16(unit) patch and accumulates all 4 gates with
// v_wmma_f32_16x16x32_bf16, then applies the LSTM cell locally.
__global__ __launch_bounds__(256) void lstm_step_kernel(
    const float*  __restrict__ xf,      // fp32 x or nullptr
    const __bf16* __restrict__ xb,      // bf16 x or nullptr
    int I,                              // input width: 64 or 256
    const __bf16* __restrict__ Wih,     // (4H, I)  bf16
    const __bf16* __restrict__ Whh,     // (4H, H)  bf16
    const float*  __restrict__ bias,    // (4H) fp32  (b_ih + b_hh)
    const __bf16* __restrict__ hin,     // (B, H) bf16
    float*        __restrict__ c,       // (B, H) fp32, in/out
    __bf16*       __restrict__ hout,    // (B, H) bf16
    float*        __restrict__ hout_f32)// nullptr, or (B,H) fp32 (final latent)
{
  const int lane = threadIdx.x & 31;
  const int w    = threadIdx.x >> 5;
  const int wm   = w & 3;            // 4 sub-tiles along batch
  const int wn   = w >> 2;           // 2 sub-tiles along units
  const int half = lane >> 4;
  const int nl   = lane & 15;

  const int m0 = blockIdx.x * 64 + wm * 16;  // batch row base of wave patch
  const int j0 = blockIdx.y * 32 + wn * 16;  // hidden-unit base of wave patch

  v8f acc[4] = {};                           // i, f, g, o accumulators (f32)

  // ---- x @ W_ih^T ----
  for (int k0 = 0; k0 < I; k0 += 32) {
    v16bf a = xb ? load_fragA_bf (xb, m0 + nl, I, k0, half)
                 : load_fragA_f32(xf, m0 + nl, I, k0, half);
    #pragma unroll
    for (int g = 0; g < 4; ++g) {
      v16bf b = load_fragB_bf(Wih, g * HID + j0 + nl, I, k0, half);
      acc[g] = __builtin_amdgcn_wmma_f32_16x16x32_bf16(
          false, a, false, b, (short)0, acc[g], false, false);
    }
  }
  // ---- h @ W_hh^T ----
  for (int k0 = 0; k0 < HID; k0 += 32) {
    v16bf a = load_fragA_bf(hin, m0 + nl, HID, k0, half);
    #pragma unroll
    for (int g = 0; g < 4; ++g) {
      v16bf b = load_fragB_bf(Whh, g * HID + j0 + nl, HID, k0, half);
      acc[g] = __builtin_amdgcn_wmma_f32_16x16x32_bf16(
          false, a, false, b, (short)0, acc[g], false, false);
    }
  }

  // ---- LSTM cell (C/D layout: lane n = unit, VGPR v + 8*half = batch row) ----
  const int j = j0 + nl;
  const float bI = bias[0 * HID + j];
  const float bF = bias[1 * HID + j];
  const float bG = bias[2 * HID + j];
  const float bO = bias[3 * HID + j];
  #pragma unroll
  for (int v = 0; v < 8; ++v) {
    const int m = m0 + v + 8 * half;
    const size_t idx = (size_t)m * HID + j;
    const float gi = sigmoidf_(acc[0][v] + bI);
    const float gf = sigmoidf_(acc[1][v] + bF);
    const float gg = tanhf_   (acc[2][v] + bG);
    const float go = sigmoidf_(acc[3][v] + bO);
    const float cn = gf * c[idx] + gi * gg;
    const float hn = go * tanhf_(cn);
    c[idx]    = cn;
    hout[idx] = f2bf(hn);
    if (hout_f32) hout_f32[idx] = hn;
  }
}

// ----------------------------- prologue utilities ----------------------------
__global__ void cvt_bf16_kernel(const float* __restrict__ src,
                                __bf16* __restrict__ dst, int n) {
  int i = blockIdx.x * blockDim.x + threadIdx.x;
  if (i < n) dst[i] = f2bf(src[i]);
}
__global__ void vec_add_kernel(const float* __restrict__ a,
                               const float* __restrict__ b,
                               float* __restrict__ d, int n) {
  int i = blockIdx.x * blockDim.x + threadIdx.x;
  if (i < n) d[i] = a[i] + b[i];
}

// ----------------------------- k-means tail ----------------------------------
// One wave per sample: argmin_k ||x - c_k||^2 (same argmin as reference's d2).
__global__ __launch_bounds__(256) void assign_kernel(
    const float* __restrict__ lat, const float* __restrict__ clusters,
    int* __restrict__ ids) {
  const int s    = blockIdx.x * 8 + (threadIdx.x >> 5);
  const int lane = threadIdx.x & 31;
  const float* x = lat + (size_t)s * HID;
  float xs[8];
  #pragma unroll
  for (int i = 0; i < 8; ++i) xs[i] = x[lane + 32 * i];
  float best = 3.4e38f; int bestk = 0;
  for (int k = 0; k < NCLUST; ++k) {
    const float* ck = clusters + (size_t)k * HID;
    float acc = 0.f;
    #pragma unroll
    for (int i = 0; i < 8; ++i) { float d = xs[i] - ck[lane + 32 * i]; acc += d * d; }
    acc = wave_reduce_sum(acc);
    acc = __shfl(acc, 0, 32);             // broadcast so all lanes agree
    if (acc < best) { best = acc; bestk = k; }  // strict < keeps first min
  }
  if (lane == 0) ids[s] = bestk;
}

// Deterministic segment mean: one thread per (cluster k, dim h); fixed-order scan.
__global__ __launch_bounds__(256) void centroid_update_kernel(
    const float* __restrict__ lat, const int* __restrict__ ids,
    const float* __restrict__ clusters, float* __restrict__ upd) {
  __shared__ int sid[BATCH];
  for (int i = threadIdx.x; i < BATCH; i += blockDim.x) sid[i] = ids[i];
  __syncthreads();
  const int k = blockIdx.x, h = threadIdx.x;
  float sum = 0.f, cnt = 0.f;
  for (int b = 0; b < BATCH; ++b)
    if (sid[b] == k) { sum += lat[(size_t)b * HID + h]; cnt += 1.f; }
  upd[(size_t)k * HID + h] =
      (cnt > 0.f) ? (sum / cnt) : clusters[(size_t)k * HID + h];
}

__global__ __launch_bounds__(256) void loss_partial_kernel(
    const float* __restrict__ lat, const float* __restrict__ upd,
    const int* __restrict__ ids, float* __restrict__ partials) {
  const int s    = blockIdx.x * 8 + (threadIdx.x >> 5);
  const int lane = threadIdx.x & 31;
  const float* x = lat + (size_t)s * HID;
  const float* u = upd + (size_t)ids[s] * HID;
  float acc = 0.f;
  #pragma unroll
  for (int i = 0; i < 8; ++i) { float d = x[lane + 32 * i] - u[lane + 32 * i]; acc += d * d; }
  acc = wave_reduce_sum(acc);
  if (lane == 0) partials[s] = acc;
}

__global__ __launch_bounds__(256) void loss_final_kernel(
    const float* __restrict__ partials, float* __restrict__ out) {
  __shared__ float sm[256];
  float a = 0.f;
  #pragma unroll
  for (int i = 0; i < 8; ++i) a += partials[threadIdx.x * 8 + i];
  sm[threadIdx.x] = a;
  __syncthreads();
  for (int s = 128; s > 0; s >>= 1) {
    if ((int)threadIdx.x < s) sm[threadIdx.x] += sm[threadIdx.x + s];
    __syncthreads();
  }
  if (threadIdx.x == 0) out[0] = 0.05f * sm[0];   // 0.5 * BETA
}

// ----------------------------- host launcher ---------------------------------
extern "C" void kernel_launch(void* const* d_in, const int* in_sizes, int n_in,
                              void* d_out, int out_size, void* d_ws, size_t ws_size,
                              hipStream_t stream) {
  const float* X     = (const float*)d_in[0];   // (200, 2048, 64)
  const float* Wih0  = (const float*)d_in[1];   // (1024, 64)
  const float* Whh0  = (const float*)d_in[2];   // (1024, 256)
  const float* bih0  = (const float*)d_in[3];
  const float* bhh0  = (const float*)d_in[4];
  const float* Wih1  = (const float*)d_in[5];   // (1024, 256)
  const float* Whh1  = (const float*)d_in[6];   // (1024, 256)
  const float* bih1  = (const float*)d_in[7];
  const float* bhh1  = (const float*)d_in[8];
  const float* clus  = (const float*)d_in[9];   // (64, 256)
  float* out = (float*)d_out;                   // latent (B*H) ++ loss (1)

  // ---- workspace carve-out (fixed part ~10 MB) ----
  char* p = (char*)d_ws;
  auto carve = [&](size_t bytes) {
    char* r = p; p += (bytes + 255) & ~(size_t)255; return r;
  };
  __bf16* wih0b = (__bf16*)carve((size_t)G4 * HIN * 2);
  __bf16* whh0b = (__bf16*)carve((size_t)G4 * HID * 2);
  __bf16* wih1b = (__bf16*)carve((size_t)G4 * HID * 2);
  __bf16* whh1b = (__bf16*)carve((size_t)G4 * HID * 2);
  float*  bias0 = (float*)carve((size_t)G4 * 4);
  float*  bias1 = (float*)carve((size_t)G4 * 4);
  __bf16* h0[2] = { (__bf16*)carve((size_t)BATCH * HID * 2),
                    (__bf16*)carve((size_t)BATCH * HID * 2) };
  __bf16* h1[2] = { (__bf16*)carve((size_t)BATCH * HID * 2),
                    (__bf16*)carve((size_t)BATCH * HID * 2) };
  float*  c0    = (float*)carve((size_t)BATCH * HID * 4);
  float*  c1    = (float*)carve((size_t)BATCH * HID * 4);
  int*    ids   = (int*)carve((size_t)BATCH * 4);
  float*  upd   = (float*)carve((size_t)NCLUST * HID * 4);
  float*  parts = (float*)carve((size_t)BATCH * 4);

  // Optional: whole-sequence bf16 copy of X (52.4 MB) to strip fp32->bf16
  // conversion VALU out of the 200-step critical path. Gated on ws_size
  // (deterministic: ws_size is fixed for the run).
  const size_t XN = (size_t)LSEQ * BATCH * HIN;
  const bool use_xbf = ((size_t)(p - (char*)d_ws) + XN * 2 <= ws_size);
  __bf16* xbf = use_xbf ? (__bf16*)carve(XN * 2) : nullptr;

  // ---- prologue: bf16 weights, fused bias, zero states ----
  auto cvt = [&](const float* s, __bf16* d, size_t n) {
    cvt_bf16_kernel<<<(int)((n + 255) / 256), 256, 0, stream>>>(s, d, (int)n);
  };
  cvt(Wih0, wih0b, (size_t)G4 * HIN);
  cvt(Whh0, whh0b, (size_t)G4 * HID);
  cvt(Wih1, wih1b, (size_t)G4 * HID);
  cvt(Whh1, whh1b, (size_t)G4 * HID);
  if (use_xbf) cvt(X, xbf, XN);
  vec_add_kernel<<<(G4 + 255) / 256, 256, 0, stream>>>(bih0, bhh0, bias0, G4);
  vec_add_kernel<<<(G4 + 255) / 256, 256, 0, stream>>>(bih1, bhh1, bias1, G4);
  hipMemsetAsync(h0[0], 0, (size_t)BATCH * HID * 2, stream);
  hipMemsetAsync(h1[0], 0, (size_t)BATCH * HID * 2, stream);
  hipMemsetAsync(c0,    0, (size_t)BATCH * HID * 4, stream);
  hipMemsetAsync(c1,    0, (size_t)BATCH * HID * 4, stream);

  // ---- 200 timesteps, two fused LSTM-step kernels each ----
  const dim3 sgrid(BATCH / 64, HID / 32);   // (32, 8)
  for (int t = 0; t < LSEQ; ++t) {
    const float*  xtf = use_xbf ? nullptr : (X + (size_t)t * BATCH * HIN);
    const __bf16* xtb = use_xbf ? (xbf + (size_t)t * BATCH * HIN) : nullptr;
    lstm_step_kernel<<<sgrid, 256, 0, stream>>>(
        xtf, xtb, HIN, wih0b, whh0b, bias0,
        h0[t & 1], c0, h0[(t + 1) & 1], nullptr);
    lstm_step_kernel<<<sgrid, 256, 0, stream>>>(
        nullptr, h0[(t + 1) & 1], HID, wih1b, whh1b, bias1,
        h1[t & 1], c1, h1[(t + 1) & 1],
        (t == LSEQ - 1) ? out : nullptr);   // final h -> latent (fp32, d_out)
  }

  // ---- k-means tail (deterministic, fp32) ----
  assign_kernel         <<<BATCH / 8, 256, 0, stream>>>(out, clus, ids);
  centroid_update_kernel<<<NCLUST,    256, 0, stream>>>(out, ids, clus, upd);
  loss_partial_kernel   <<<BATCH / 8, 256, 0, stream>>>(out, upd, ids, parts);
  loss_final_kernel     <<<1,         256, 0, stream>>>(parts, out + (size_t)BATCH * HID);
}
